// KnowledgeGraphGAT_68264210202756
// MI455X (gfx1250) — matrix operand order repaired and checked by reading, not compiled, not run on hardware
//
#include <hip/hip_runtime.h>

#define NE   4096
#define DD   128
#define HH   4
#define NMID 9
#define LRA  0.2f

typedef __attribute__((ext_vector_type(16))) __bf16 v16bf;
typedef __attribute__((ext_vector_type(8)))  float  v8f;

union Frag {
  v16bf v;
  unsigned short u[16];
  uint4 q[2];
};

__device__ __forceinline__ unsigned short f2bf(float f) {
  unsigned u = __float_as_uint(f);
  u += 0x7FFFu + ((u >> 16) & 1u);
  return (unsigned short)(u >> 16);
}

// ---- pack adjacency into bitmask: bits[row][w] bit k = adj[row][w*32+k] > 0 ----
__global__ __launch_bounds__(32) void pack_mask_kernel(const float* __restrict__ adj,
                                                       unsigned* __restrict__ bits) {
  int row = blockIdx.x;
  int lane = threadIdx.x;
  for (int w = 0; w < NE / 32; ++w) {
    float v = adj[(size_t)row * NE + w * 32 + lane];
    unsigned long long b = __ballot(v > 0.0f);
    if (lane == 0) bits[row * (NE / 32) + w] = (unsigned)b;
  }
}

// ---- pack [K,128] (per head) f32 matrix into WMMA B-fragment bf16 layout ----
// elem (((c*8 + t)*32 + lane)*16 + p) <- src[k=c*32+kb+(p<8?p:8+p), kb=(lane<16?0:8)][col=t*16+lane%16]
__global__ void pack_b_kernel(const float* __restrict__ src, unsigned short* __restrict__ dst,
                              int K, int nHeads) {
  long headElems = (long)K * 128;
  long total = nHeads * headElems;
  for (long idx = blockIdx.x * (long)blockDim.x + threadIdx.x; idx < total;
       idx += (long)gridDim.x * blockDim.x) {
    int h = (int)(idx / headElems);
    int e = (int)(idx % headElems);
    int c = e >> 12;            // /(8*32*16)
    int rem = e & 4095;
    int t = rem >> 9;
    int l = (rem >> 4) & 31;
    int p = rem & 15;
    int k = c * 32 + ((p < 8) ? (((l < 16) ? 0 : 8) + p) : (8 + ((l < 16) ? 0 : 8) + p));
    int col = t * 16 + (l & 15);
    dst[idx] = f2bf(src[(size_t)h * headElems + (size_t)k * 128 + col]);
  }
}

// ---- pack x [4096, Fin] f32 into WMMA A-fragment bf16 layout ----
__global__ void pack_a_kernel(const float* __restrict__ src, unsigned short* __restrict__ dst,
                              int Fin) {
  int nch = Fin / 32;
  long total = (long)NE * Fin;
  int per = nch * 512;          // elems per 16-row tile
  for (long idx = blockIdx.x * (long)blockDim.x + threadIdx.x; idx < total;
       idx += (long)gridDim.x * blockDim.x) {
    int R = (int)(idx / per);
    int rem = (int)(idx % per);
    int c = rem >> 9;
    int l = (rem >> 4) & 31;
    int p = rem & 15;
    int m = l & 15;
    int kb = (l < 16) ? 0 : 8;
    int k = (p < 8) ? (kb + p) : (8 + kb + p);   // K = {kb..kb+7, 16+kb..23+kb}
    dst[idx] = f2bf(src[(size_t)(R * 16 + m) * Fin + c * 32 + k]);
  }
}

__device__ __forceinline__ void loadB4(const unsigned short* __restrict__ base,
                                       int c, int t0, int lane, Frag* b) {
#pragma unroll
  for (int i = 0; i < 4; ++i) {
    const uint4* bp = (const uint4*)(base + ((size_t)(c * 8 + t0 + i) * 32 + lane) * 16);
    b[i].q[0] = bp[0];
    b[i].q[1] = bp[1];
  }
}

__device__ __forceinline__ void loadA(const unsigned short* __restrict__ xb,
                                      int R, int nch, int c, int lane, Frag& a) {
  const uint4* ap = (const uint4*)(xb + ((size_t)(R * nch + c) * 32 + lane) * 16);
  a.q[0] = ap[0];
  a.q[1] = ap[1];
}

// ---- Wh = x @ W per head via WMMA (bf16 in, f32 out), software-pipelined ----
__global__ __launch_bounds__(32) void xw_gemm_kernel(const unsigned short* __restrict__ xb,
                                                     const unsigned short* __restrict__ wb,
                                                     float* __restrict__ wh,
                                                     int nch, int wbHeadElems) {
  int R = blockIdx.x, h = blockIdx.y, lane = threadIdx.x;
  const unsigned short* wbh = wb + (size_t)h * wbHeadElems;
  const v8f vz = {0.f, 0.f, 0.f, 0.f, 0.f, 0.f, 0.f, 0.f};
  v8f acc[8];
#pragma unroll
  for (int t = 0; t < 8; ++t) acc[t] = vz;

  Frag a, an;
  Frag bA[4], bB[4];
  loadA(xb, R, nch, 0, lane, a);
  loadB4(wbh, 0, 0, lane, bA);
  for (int c = 0; c < nch; ++c) {
    loadB4(wbh, c, 4, lane, bB);                   // tiles 4-7 of this chunk
    if (c + 1 < nch) loadA(xb, R, nch, c + 1, lane, an);
#pragma unroll
    for (int i = 0; i < 4; ++i)                    // overlap with bB loads
      acc[i] = __builtin_amdgcn_wmma_f32_16x16x32_bf16(false, a.v, false, bA[i].v,
                                                       (short)0, acc[i], false, false);
    if (c + 1 < nch) loadB4(wbh, c + 1, 0, lane, bA);  // tiles 0-3 of next chunk
#pragma unroll
    for (int i = 0; i < 4; ++i)                    // overlap with bA loads
      acc[4 + i] = __builtin_amdgcn_wmma_f32_16x16x32_bf16(false, a.v, false, bB[i].v,
                                                           (short)0, acc[4 + i], false, false);
    if (c + 1 < nch) a = an;
  }

  int rbase = R * 16 + ((lane < 16) ? 0 : 8);
  int col0 = lane & 15;
  float* out = wh + (size_t)h * NE * 128;
#pragma unroll
  for (int t = 0; t < 8; ++t)
#pragma unroll
    for (int r = 0; r < 8; ++r)
      out[(size_t)(rbase + r) * 128 + t * 16 + col0] = acc[t][r];
}

// ---- f1 = Wh@a[:128], f2 = Wh@a[128:] ----
__global__ __launch_bounds__(32) void compute_f_kernel(const float* __restrict__ wh,
                                                       const float* __restrict__ avec,
                                                       float* __restrict__ f1,
                                                       float* __restrict__ f2) {
  int row = blockIdx.x, h = blockIdx.y, lane = threadIdx.x;
  const float* w = wh + ((size_t)h * NE + row) * 128;
  const float* a = avec + h * 256;
  float s1 = 0.f, s2 = 0.f;
  for (int i = lane; i < 128; i += 32) {
    float v = w[i];
    s1 += v * a[i];
    s2 += v * a[128 + i];
  }
  for (int off = 16; off > 0; off >>= 1) {
    s1 += __shfl_xor(s1, off, 32);
    s2 += __shfl_xor(s2, off, 32);
  }
  if (lane == 0) { f1[h * NE + row] = s1; f2[h * NE + row] = s2; }
}

// ---- per-row softmax stats: mx (row max of masked leakyrelu), rinv = 1/sum(exp) ----
__global__ __launch_bounds__(32) void softmax_stats_kernel(const unsigned* __restrict__ bits,
                                                           const float* __restrict__ f1,
                                                           const float* __restrict__ f2,
                                                           float* __restrict__ mx,
                                                           float* __restrict__ rinv) {
  int row = blockIdx.x, h = blockIdx.y, lane = threadIdx.x;
  float f1r = f1[h * NE + row];
  const float* f2h = f2 + h * NE;
  const unsigned* br = bits + row * (NE / 32);
  float m = -3.0e38f;
  for (int w = 0; w < NE / 32; ++w) {
    unsigned bm = br[w];
    if ((bm >> lane) & 1u) {
      float e = f1r + f2h[w * 32 + lane];
      e = (e > 0.f) ? e : LRA * e;
      m = fmaxf(m, e);
    }
  }
  for (int off = 16; off > 0; off >>= 1) m = fmaxf(m, __shfl_xor(m, off, 32));
  float s = 0.f;
  for (int w = 0; w < NE / 32; ++w) {
    unsigned bm = br[w];
    if ((bm >> lane) & 1u) {
      float e = f1r + f2h[w * 32 + lane];
      e = (e > 0.f) ? e : LRA * e;
      s += __expf(e - m);
    }
  }
  for (int off = 16; off > 0; off >>= 1) s += __shfl_xor(s, off, 32);
  if (lane == 0) { mx[h * NE + row] = m; rinv[h * NE + row] = 1.0f / s; }
}

// ---- out = elu( softmax(e) @ Wh ): B loads issued first, P built while in flight ----
__global__ __launch_bounds__(32) void attn_gemm_kernel(const unsigned* __restrict__ bits,
                                                       const float* __restrict__ f1,
                                                       const float* __restrict__ f2,
                                                       const float* __restrict__ mx,
                                                       const float* __restrict__ rinv,
                                                       const unsigned short* __restrict__ whb,
                                                       float* __restrict__ out, int ldo) {
  int R = blockIdx.x, h = blockIdx.y, lane = threadIdx.x;
  int m0 = lane & 15;
  int rowA = R * 16 + m0;
  float f1r = f1[h * NE + rowA];
  float mxr = mx[h * NE + rowA];
  float rir = rinv[h * NE + rowA];
  const float* f2h = f2 + h * NE;
  const unsigned short* whh = whb + (size_t)h * NE * 128;
  int kb = (lane < 16) ? 0 : 8;
  const v8f vz = {0.f, 0.f, 0.f, 0.f, 0.f, 0.f, 0.f, 0.f};
  v8f acc[8];
#pragma unroll
  for (int t = 0; t < 8; ++t) acc[t] = vz;

  unsigned bm = bits[rowA * (NE / 32)];
  float f2v = f2h[lane];
  for (int c = 0; c < NE / 32; ++c) {
    // 1) issue all B-fragment loads for this chunk (one big clause)
    Frag b[8];
#pragma unroll
    for (int t = 0; t < 8; ++t) {
      const uint4* bp = (const uint4*)(whh + ((size_t)(c * 8 + t) * 32 + lane) * 16);
      b[t].q[0] = bp[0];
      b[t].q[1] = bp[1];
    }
    // 2) prefetch next chunk's mask word + f2 column value
    unsigned bmn = 0u;
    float f2vn = 0.f;
    if (c + 1 < NE / 32) {
      bmn = bits[rowA * (NE / 32) + c + 1];
      f2vn = f2h[(c + 1) * 32 + lane];
    }
    // 3) build P fragment (VALU/trans work hides the B-load latency)
    Frag a;
#pragma unroll
    for (int p = 0; p < 16; ++p) {
      int k = (p < 8) ? (kb + p) : (8 + kb + p);   // A-frag K mapping (ISA 7.12.2)
      float fv = __shfl(f2v, k, 32);
      float e = f1r + fv;
      e = (e > 0.f) ? e : LRA * e;
      float pv = ((bm >> k) & 1u) ? __expf(e - mxr) * rir : 0.0f;
      a.u[p] = f2bf(pv);
    }
    // 4) back-to-back WMMAs
#pragma unroll
    for (int t = 0; t < 8; ++t)
      acc[t] = __builtin_amdgcn_wmma_f32_16x16x32_bf16(false, a.v, false, b[t].v,
                                                       (short)0, acc[t], false, false);
    bm = bmn;
    f2v = f2vn;
  }

  int rbase = R * 16 + ((lane < 16) ? 0 : 8);
  int col0 = (lane & 15) + h * 128;   // ldo==512 middle layers; ldo==128 final (h==0)
#pragma unroll
  for (int t = 0; t < 8; ++t)
#pragma unroll
    for (int r = 0; r < 8; ++r) {
      float v = acc[t][r];
      v = (v > 0.f) ? v : (__expf(v) - 1.0f);       // elu
      out[(size_t)(rbase + r) * ldo + col0 + t * 16] = v;
    }
}

// ---- head_relation_combined = entity_emb[head] + relation_emb[relation] ----
__global__ void head_rel_kernel(const int* __restrict__ head, const int* __restrict__ rel,
                                const float* __restrict__ ent, const float* __restrict__ relemb,
                                float* __restrict__ out) {
  int idx = blockIdx.x * blockDim.x + threadIdx.x;
  if (idx >= NE * DD) return;
  int i = idx >> 7, c = idx & 127;
  out[idx] = ent[(size_t)head[i] * DD + c] + relemb[(size_t)rel[i] * DD + c];
}

extern "C" void kernel_launch(void* const* d_in, const int* in_sizes, int n_in,
                              void* d_out, int out_size, void* d_ws, size_t ws_size,
                              hipStream_t stream) {
  const int*   head = (const int*)d_in[0];
  const int*   rel  = (const int*)d_in[1];
  const float* adj  = (const float*)d_in[2];
  const float* ent  = (const float*)d_in[3];
  const float* rele = (const float*)d_in[4];
  const float* W0   = (const float*)d_in[5];
  const float* a0   = (const float*)d_in[6];
  const float* Wm   = (const float*)d_in[7];
  const float* am   = (const float*)d_in[8];
  const float* Wo   = (const float*)d_in[9];
  const float* ao   = (const float*)d_in[10];
  float* out = (float*)d_out;

  // workspace carve-up
  char* w = (char*)d_ws;
  auto take = [&](size_t bytes) -> char* {
    char* p = w;
    w += (bytes + 255) & ~(size_t)255;
    return p;
  };
  unsigned*       maskbits = (unsigned*)      take((size_t)NE * (NE / 32) * 4);   // 2 MB
  unsigned short* xb       = (unsigned short*)take((size_t)NE * 512 * 2);         // 4 MB
  float*          Wh       = (float*)         take((size_t)HH * NE * 128 * 4);    // 8 MB
  unsigned short* Whb      = (unsigned short*)take((size_t)HH * NE * 128 * 2);    // 4 MB
  float*          f1       = (float*)         take((size_t)HH * NE * 4);
  float*          f2       = (float*)         take((size_t)HH * NE * 4);
  float*          mxv      = (float*)         take((size_t)HH * NE * 4);
  float*          riv      = (float*)         take((size_t)HH * NE * 4);
  float*          xA       = (float*)         take((size_t)NE * 512 * 4);         // 8 MB
  float*          xB       = (float*)         take((size_t)NE * 512 * 4);         // 8 MB
  unsigned short* W0b      = (unsigned short*)take((size_t)HH * 128 * 128 * 2);
  unsigned short* Wmb      = (unsigned short*)take((size_t)NMID * HH * 512 * 128 * 2);
  unsigned short* Wob      = (unsigned short*)take((size_t)512 * 128 * 2);
  (void)ws_size; (void)in_sizes; (void)n_in; (void)out_size;

  // one-time packing (deterministic per call)
  pack_mask_kernel<<<NE, 32, 0, stream>>>(adj, maskbits);
  pack_b_kernel<<<512, 256, 0, stream>>>(W0, W0b, 128, HH);
  pack_b_kernel<<<4096, 256, 0, stream>>>(Wm, Wmb, 512, NMID * HH);
  pack_b_kernel<<<512, 256, 0, stream>>>(Wo, Wob, 512, 1);
  head_rel_kernel<<<(NE * DD + 255) / 256, 256, 0, stream>>>(head, rel, ent, rele,
                                                             out + (size_t)NE * DD);

  const int wbHead128 = (128 / 32) * 8 * 32 * 16;  // 16384 elems per head (Fin=128)
  const int wbHead512 = (512 / 32) * 8 * 32 * 16;  // 65536 elems per head (Fin=512)

  // ---- layer 0: entity_emb (Fin=128) -> xA [4096, 512] ----
  pack_a_kernel<<<2048, 256, 0, stream>>>(ent, xb, 128);
  xw_gemm_kernel<<<dim3(256, HH), 32, 0, stream>>>(xb, W0b, Wh, 128 / 32, wbHead128);
  compute_f_kernel<<<dim3(NE, HH), 32, 0, stream>>>(Wh, a0, f1, f2);
  pack_b_kernel<<<8192, 256, 0, stream>>>(Wh, Whb, NE, HH);
  softmax_stats_kernel<<<dim3(NE, HH), 32, 0, stream>>>(maskbits, f1, f2, mxv, riv);
  attn_gemm_kernel<<<dim3(256, HH), 32, 0, stream>>>(maskbits, f1, f2, mxv, riv, Whb, xA, 512);

  // ---- 9 middle layers (Fin=512) ----
  float* cur = xA;
  float* nxt = xB;
  for (int L = 0; L < NMID; ++L) {
    pack_a_kernel<<<8192, 256, 0, stream>>>(cur, xb, 512);
    xw_gemm_kernel<<<dim3(256, HH), 32, 0, stream>>>(xb, Wmb + (size_t)L * HH * wbHead512,
                                                     Wh, 512 / 32, wbHead512);
    compute_f_kernel<<<dim3(NE, HH), 32, 0, stream>>>(Wh, am + (size_t)L * HH * 256, f1, f2);
    pack_b_kernel<<<8192, 256, 0, stream>>>(Wh, Whb, NE, HH);
    softmax_stats_kernel<<<dim3(NE, HH), 32, 0, stream>>>(maskbits, f1, f2, mxv, riv);
    attn_gemm_kernel<<<dim3(256, HH), 32, 0, stream>>>(maskbits, f1, f2, mxv, riv, Whb, nxt, 512);
    float* tmp = cur; cur = nxt; nxt = tmp;
  }

  // ---- output layer: single head, Fin=512 -> elu -> d_out[0 : 4096*128] ----
  pack_a_kernel<<<8192, 256, 0, stream>>>(cur, xb, 512);
  xw_gemm_kernel<<<dim3(256, 1), 32, 0, stream>>>(xb, Wob, Wh, 512 / 32, wbHead512);
  compute_f_kernel<<<dim3(NE, 1), 32, 0, stream>>>(Wh, ao, f1, f2);
  pack_b_kernel<<<2048, 256, 0, stream>>>(Wh, Whb, NE, 1);
  softmax_stats_kernel<<<dim3(NE, 1), 32, 0, stream>>>(maskbits, f1, f2, mxv, riv);
  attn_gemm_kernel<<<dim3(256, 1), 32, 0, stream>>>(maskbits, f1, f2, mxv, riv, Whb, out, 128);
}
